// Model_39281770889443
// MI455X (gfx1250) — compile-verified, matrix operand order checked by reading, not run on hardware
//
#include <hip/hip_runtime.h>
#include <hip/hip_bf16.h>
#include <math.h>

#define TT 2048
#define VV 32000
#define DD 1024
#define EE 16
#define CAPP 256
#define DFFF 2048
#define EPSS 1e-5f
#define APAD 16          // bf16 elements of row padding in LDS A tiles (32B)

typedef __attribute__((ext_vector_type(16))) __bf16 v16bf;
typedef __attribute__((ext_vector_type(8)))  float  v8f;
typedef __attribute__((ext_vector_type(4)))  int    v4i_;

#if defined(__has_builtin)
#  if __has_builtin(__builtin_amdgcn_global_load_async_to_lds_b128)
#    define HAS_ASYNC_LDS 1
#  endif
#endif

#define WMMA_BF16(a, b, c) \
    __builtin_amdgcn_wmma_f32_16x16x32_bf16(false, (a), false, (b), (short)0, (c), false, false)

// ---------------------------------------------------------------------------
// fp32 -> bf16 elementwise convert (used for embed table)
// ---------------------------------------------------------------------------
__global__ void convert_bf16_kernel(const float* __restrict__ in, __bf16* __restrict__ out, long n) {
    long i = ((long)blockIdx.x * blockDim.x + threadIdx.x) * 4;
    if (i + 3 < n) {
        out[i+0] = (__bf16)in[i+0];
        out[i+1] = (__bf16)in[i+1];
        out[i+2] = (__bf16)in[i+2];
        out[i+3] = (__bf16)in[i+3];
    } else {
        for (long j = i; j < n; ++j) out[j] = (__bf16)in[j];
    }
}

// ---------------------------------------------------------------------------
// Embedding gather: h[t,:] = embed_W[ids[t],:]
// ---------------------------------------------------------------------------
__global__ void embed_gather_kernel(const int* __restrict__ ids, const float* __restrict__ embed,
                                    float* __restrict__ h) {
    int t = blockIdx.x;
    const float* src = embed + (size_t)ids[t] * DD;
    float* dst = h + (size_t)t * DD;
    for (int d = threadIdx.x; d < DD; d += blockDim.x) dst[d] = src[d];
}

// ---------------------------------------------------------------------------
// Router: logits = h @ Wr ; softmax ; hard top-2 (ties -> lower index, like lax.top_k)
// ---------------------------------------------------------------------------
__global__ void router_kernel(const float* __restrict__ h, const float* __restrict__ Wr,
                              float* __restrict__ logits, float* __restrict__ probs,
                              unsigned* __restrict__ maskbits, int* __restrict__ top2) {
    int t = blockIdx.x * blockDim.x + threadIdx.x;
    if (t >= TT) return;
    float acc[EE];
#pragma unroll
    for (int e = 0; e < EE; ++e) acc[e] = 0.f;
    const float* hr = h + (size_t)t * DD;
    for (int d = 0; d < DD; ++d) {
        float hv = hr[d];
        const float* w = Wr + (size_t)d * EE;
#pragma unroll
        for (int e = 0; e < EE; ++e) acc[e] = fmaf(hv, w[e], acc[e]);
    }
    float mx = acc[0];
#pragma unroll
    for (int e = 1; e < EE; ++e) mx = fmaxf(mx, acc[e]);
    float p[EE]; float s = 0.f;
#pragma unroll
    for (int e = 0; e < EE; ++e) { p[e] = __expf(acc[e] - mx); s += p[e]; }
    float inv = 1.f / s;
#pragma unroll
    for (int e = 0; e < EE; ++e) {
        logits[(size_t)t * EE + e] = acc[e];
        probs [(size_t)t * EE + e] = p[e] * inv;
    }
    int b0 = 0;
#pragma unroll
    for (int e = 1; e < EE; ++e) if (acc[e] > acc[b0]) b0 = e;
    int b1 = (b0 == 0) ? 1 : 0;
#pragma unroll
    for (int e = 0; e < EE; ++e) if (e != b0 && acc[e] > acc[b1]) b1 = e;
    top2[2 * t + 0] = b0;
    top2[2 * t + 1] = b1;
    maskbits[t] = (1u << b0) | (1u << b1);
}

// ---------------------------------------------------------------------------
// Capacity: per expert, keep top-CAP assigned tokens ranked by logit
// ---------------------------------------------------------------------------
__global__ void cap_rank_kernel(const float* __restrict__ logits, const unsigned* __restrict__ maskbits,
                                int* __restrict__ kept) {
    int t = blockIdx.x * blockDim.x + threadIdx.x;
    int e = blockIdx.y;
    if (t >= TT) return;
    int keep = 0;
    if ((maskbits[t] >> e) & 1u) {
        float myl = logits[(size_t)t * EE + e];
        int rank = 0;
        for (int tt = 0; tt < TT; ++tt) {
            if ((maskbits[tt] >> e) & 1u) {
                float lq = logits[(size_t)tt * EE + e];
                rank += (lq > myl) || (lq == myl && tt < t);
            }
        }
        keep = (rank < CAPP);
    }
    kept[(size_t)e * TT + t] = keep;
}

// Position-ordered compaction of kept tokens into slots (== stable sort by pos)
__global__ void cap_prefix_kernel(const int* __restrict__ kept, int* __restrict__ slot_of,
                                  int* __restrict__ tok_of, int* __restrict__ nslots) {
    int e = blockIdx.x;
    if (threadIdx.x != 0) return;
    int run = 0;
    for (int t = 0; t < TT; ++t) {
        if (kept[(size_t)e * TT + t] && run < CAPP) {
            slot_of[(size_t)e * TT + t] = run;
            tok_of[(size_t)e * CAPP + run] = t;
            run++;
        }
    }
    nslots[e] = run;
}

// Gather token rows into expert slot buffers, fp32 -> bf16
__global__ void gather_kernel(const float* __restrict__ h, const int* __restrict__ tok_of,
                              const int* __restrict__ nslots, __bf16* __restrict__ xin) {
    int ec = blockIdx.x;            // e*CAP + c
    int e = ec / CAPP, c = ec % CAPP;
    __bf16* dst = xin + (size_t)ec * DD;
    if (c < nslots[e]) {
        const float* src = h + (size_t)tok_of[ec] * DD;
        for (int d = threadIdx.x; d < DD; d += blockDim.x) dst[d] = (__bf16)src[d];
    } else {
        for (int d = threadIdx.x; d < DD; d += blockDim.x) dst[d] = (__bf16)0.f;
    }
}

// ---------------------------------------------------------------------------
// Tiled transpose + fp32->bf16 convert: in (R,S) row-major -> out (S,R)
// ---------------------------------------------------------------------------
__global__ void transpose_convert_kernel(const float* __restrict__ in, __bf16* __restrict__ out,
                                         int R, int S) {
    __shared__ float tile[32][33];
    const float* inp = in + (size_t)blockIdx.z * R * S;
    __bf16* outp = out + (size_t)blockIdx.z * R * S;
    int r0 = blockIdx.x * 32, s0 = blockIdx.y * 32;
    for (int i = threadIdx.y; i < 32; i += 8)
        tile[i][threadIdx.x] = inp[(size_t)(r0 + i) * S + s0 + threadIdx.x];
    __syncthreads();
    for (int i = threadIdx.y; i < 32; i += 8)
        outp[(size_t)(s0 + i) * R + r0 + threadIdx.x] = (__bf16)tile[threadIdx.x][i];
}

// ---------------------------------------------------------------------------
// Cooperative stage of a 16 x K bf16 A-tile into LDS via async global->LDS DMA
// (ASYNCcnt path on CDNA5); falls back to register copies if builtin absent.
// Builtin signature (from toolchain): (global int4* src, local int4* dst,
// imm offset, imm cpol).
// ---------------------------------------------------------------------------
__device__ __forceinline__ void stage_a_tile(const __bf16* __restrict__ Ag, int ldg,
                                             __bf16* As, int lds_stride, int K, int nthreads) {
    const int c8 = K / 8;      // 16-byte chunks per row
#ifdef HAS_ASYNC_LDS
    for (int idx = threadIdx.x; idx < 16 * c8; idx += nthreads) {
        int row = idx / c8;
        int col = (idx - row * c8) * 8;
        __builtin_amdgcn_global_load_async_to_lds_b128(
            (__attribute__((address_space(1))) v4i_*)(Ag + (size_t)row * ldg + col),
            (__attribute__((address_space(3))) v4i_*)(As + (size_t)row * lds_stride + col),
            0, 0);
    }
#  if __has_builtin(__builtin_amdgcn_s_wait_asynccnt)
    __builtin_amdgcn_s_wait_asynccnt(0);
#  else
    asm volatile("s_wait_asynccnt 0x0" ::: "memory");
#  endif
#else
    for (int idx = threadIdx.x; idx < 16 * c8; idx += nthreads) {
        int row = idx / c8;
        int col = (idx - row * c8) * 8;
        *(uint4*)(As + (size_t)row * lds_stride + col) =
            *(const uint4*)(Ag + (size_t)row * ldg + col);
    }
#endif
    __syncthreads();
}

// ---------------------------------------------------------------------------
// Register-double-buffered 16(M) x 64(N) wave tile, K steps of 32.
// A: 16 x K (K contiguous, stride lda) -- may be an LDS or global pointer.
// B: N x K rows (K contiguous, stride ldb), global.
// Next k-tile loads are issued before the current WMMAs so the memory waits
// become partial instead of s_wait_loadcnt 0.
// ---------------------------------------------------------------------------
__device__ __forceinline__ void wmma_kloop(const __bf16* A, int lda,
                                           const __bf16* B, int ldb,
                                           int K, v8f acc[4]) {
    const int lane = threadIdx.x & 31;
    const int lr = lane & 15;
    const int lk = (lane >> 4) << 4;
    const __bf16* ap = A + (size_t)lr * lda + lk;
    const __bf16* bp = B + (size_t)lr * ldb + lk;
    const size_t bs = (size_t)16 * ldb;

    v16bf ac  = *(const v16bf*)(ap);
    v16bf bc0 = *(const v16bf*)(bp);
    v16bf bc1 = *(const v16bf*)(bp + bs);
    v16bf bc2 = *(const v16bf*)(bp + 2 * bs);
    v16bf bc3 = *(const v16bf*)(bp + 3 * bs);

    for (int k = 32; k < K; k += 32) {
        v16bf an  = *(const v16bf*)(ap + k);
        v16bf bn0 = *(const v16bf*)(bp + k);
        v16bf bn1 = *(const v16bf*)(bp + bs + k);
        v16bf bn2 = *(const v16bf*)(bp + 2 * bs + k);
        v16bf bn3 = *(const v16bf*)(bp + 3 * bs + k);
        __builtin_prefetch(bp + k + 32, 0, 3);           // WGP-scope RT prefetch
        __builtin_prefetch(bp + 3 * bs + k + 32, 0, 3);
        acc[0] = WMMA_BF16(ac, bc0, acc[0]);
        acc[1] = WMMA_BF16(ac, bc1, acc[1]);
        acc[2] = WMMA_BF16(ac, bc2, acc[2]);
        acc[3] = WMMA_BF16(ac, bc3, acc[3]);
        ac = an; bc0 = bn0; bc1 = bn1; bc2 = bn2; bc3 = bn3;
    }
    acc[0] = WMMA_BF16(ac, bc0, acc[0]);
    acc[1] = WMMA_BF16(ac, bc1, acc[1]);
    acc[2] = WMMA_BF16(ac, bc2, acc[2]);
    acc[3] = WMMA_BF16(ac, bc3, acc[3]);
}

__device__ __forceinline__ float gelu_tanh(float x) {
    return 0.5f * x * (1.f + tanhf(0.7978845608028654f * (x + 0.044715f * x * x * x)));
}

// ---------------------------------------------------------------------------
// Expert GEMM1 + GELU: h1[e] = gelu(xin[e] (C x D) @ W1[e]) -> (C x DFF), bf16.
// A tile (16 x 1024) staged once per block in LDS via async DMA; all 4 waves
// read it through ds_load while B streams from global.
// ---------------------------------------------------------------------------
__global__ void gemm1_gelu_kernel(const __bf16* __restrict__ xin, const __bf16* __restrict__ w1t,
                                  __bf16* __restrict__ h1) {
    __shared__ __bf16 As[16 * (DD + APAD)];
    int e = blockIdx.z;
    int m0 = blockIdx.x * 16;
    int wave = threadIdx.x >> 5;
    int n0 = blockIdx.y * 256 + wave * 64;

    stage_a_tile(xin + (size_t)e * CAPP * DD + (size_t)m0 * DD, DD, As, DD + APAD, DD, 128);

    const __bf16* B = w1t + (size_t)e * DFFF * DD + (size_t)n0 * DD;
    v8f acc[4] = {};
    wmma_kloop(As, DD + APAD, B, DD, DD, acc);

    int lane = threadIdx.x & 31;
    int mrow = m0 + 8 * (lane >> 4);
    int ncol = lane & 15;
    __bf16* out = h1 + (size_t)e * CAPP * DFFF;
#pragma unroll
    for (int s = 0; s < 4; ++s)
#pragma unroll
        for (int r = 0; r < 8; ++r)
            out[(size_t)(mrow + r) * DFFF + n0 + s * 16 + ncol] = (__bf16)gelu_tanh(acc[s][r]);
}

// Expert GEMM2: eout[e] = h1[e] (C x DFF) @ W2[e] -> (C x D), fp32 out.
// K=2048: keep A in registers (double-buffered) to stay under LDS limits.
__global__ void gemm2_kernel(const __bf16* __restrict__ h1, const __bf16* __restrict__ w2t,
                             float* __restrict__ eout) {
    int e = blockIdx.z;
    int m0 = blockIdx.x * 16;
    int wave = threadIdx.x >> 5;
    int n0 = blockIdx.y * 256 + wave * 64;
    const __bf16* A = h1 + (size_t)e * CAPP * DFFF + (size_t)m0 * DFFF;
    const __bf16* B = w2t + (size_t)e * DD * DFFF + (size_t)n0 * DFFF;
    v8f acc[4] = {};
    wmma_kloop(A, DFFF, B, DFFF, DFFF, acc);
    int lane = threadIdx.x & 31;
    int mrow = m0 + 8 * (lane >> 4);
    int ncol = lane & 15;
    float* out = eout + (size_t)e * CAPP * DD;
#pragma unroll
    for (int s = 0; s < 4; ++s)
#pragma unroll
        for (int r = 0; r < 8; ++r)
            out[(size_t)(mrow + r) * DD + n0 + s * 16 + ncol] = acc[s][r];
}

// Combine: h_next = (1 - rho)*h + sum_e cw_e * expert_out[e, slot]
__global__ void combine_kernel(const float* __restrict__ hin, const float* __restrict__ probs,
                               const int* __restrict__ top2, const int* __restrict__ kept,
                               const int* __restrict__ slot_of, const float* __restrict__ eout,
                               float* __restrict__ hout) {
    int t = blockIdx.x;
    int e0 = top2[2 * t + 0], e1 = top2[2 * t + 1];
    int k0 = kept[(size_t)e0 * TT + t];
    int k1 = kept[(size_t)e1 * TT + t];
    float c0 = k0 ? probs[(size_t)t * EE + e0] : 0.f;
    float c1 = k1 ? probs[(size_t)t * EE + e1] : 0.f;
    float onemr = 1.f - (c0 + c1);
    const float* o0 = eout + (k0 ? ((size_t)e0 * CAPP + slot_of[(size_t)e0 * TT + t]) * DD : 0);
    const float* o1 = eout + (k1 ? ((size_t)e1 * CAPP + slot_of[(size_t)e1 * TT + t]) * DD : 0);
    const float* hr = hin + (size_t)t * DD;
    float* ho = hout + (size_t)t * DD;
    for (int d = threadIdx.x; d < DD; d += blockDim.x)
        ho[d] = onemr * hr[d] + c0 * o0[d] + c1 * o1[d];
}

// RMSNorm + scale, fp32 -> bf16
__global__ void rmsnorm_kernel(const float* __restrict__ h, const float* __restrict__ scale,
                               __bf16* __restrict__ out) {
    __shared__ float red[256];
    int t = blockIdx.x;
    const float* hr = h + (size_t)t * DD;
    float s = 0.f;
    for (int d = threadIdx.x; d < DD; d += blockDim.x) { float v = hr[d]; s += v * v; }
    red[threadIdx.x] = s;
    __syncthreads();
    for (int w = 128; w > 0; w >>= 1) {
        if (threadIdx.x < w) red[threadIdx.x] += red[threadIdx.x + w];
        __syncthreads();
    }
    float inv = rsqrtf(red[0] * (1.f / DD) + EPSS);
    __bf16* o = out + (size_t)t * DD;
    for (int d = threadIdx.x; d < DD; d += blockDim.x)
        o[d] = (__bf16)(hr[d] * inv * scale[d]);
}

// Unembedding: out (T x V) = hn_bf (T x D) @ embed_bf^T ; embed_bf (V x D) is
// already the N x K layout WMMA's B operand wants. A tile staged in LDS.
__global__ void unembed_kernel(const __bf16* __restrict__ hn, const __bf16* __restrict__ emb,
                               float* __restrict__ out) {
    __shared__ __bf16 As[16 * (DD + APAD)];
    int m0 = blockIdx.x * 16;
    int wave = threadIdx.x >> 5;
    int n0 = blockIdx.y * 256 + wave * 64;

    stage_a_tile(hn + (size_t)m0 * DD, DD, As, DD + APAD, DD, 128);

    const __bf16* B = emb + (size_t)n0 * DD;
    v8f acc[4] = {};
    wmma_kloop(As, DD + APAD, B, DD, DD, acc);

    int lane = threadIdx.x & 31;
    int mrow = m0 + 8 * (lane >> 4);
    int ncol = lane & 15;
#pragma unroll
    for (int s = 0; s < 4; ++s)
#pragma unroll
        for (int r = 0; r < 8; ++r)
            out[(size_t)(mrow + r) * VV + n0 + s * 16 + ncol] = acc[s][r];
}

// ---------------------------------------------------------------------------
extern "C" void kernel_launch(void* const* d_in, const int* in_sizes, int n_in,
                              void* d_out, int out_size, void* d_ws, size_t ws_size,
                              hipStream_t stream) {
    const int*   ids     = (const int*)d_in[0];
    const float* embed_W = (const float*)d_in[1];
    const float* ln_s    = (const float*)d_in[2];
    const float* routerW = (const float*)d_in[3];
    const float* exW1    = (const float*)d_in[4];
    const float* exW2    = (const float*)d_in[5];
    float* out = (float*)d_out;

    char* ws = (char*)d_ws;
    size_t off = 0;
    auto alloc = [&](size_t bytes) { size_t o = off; off = (off + bytes + 255) & ~(size_t)255; return o; };

    __bf16*   embed_bf = (__bf16*)(ws + alloc((size_t)VV * DD * 2));
    float*    h_a      = (float*) (ws + alloc((size_t)TT * DD * 4));
    float*    h_b      = (float*) (ws + alloc((size_t)TT * DD * 4));
    __bf16*   hn_bf    = (__bf16*)(ws + alloc((size_t)TT * DD * 2));
    float*    logits   = (float*) (ws + alloc((size_t)TT * EE * 4));
    float*    probs    = (float*) (ws + alloc((size_t)TT * EE * 4));
    unsigned* maskbits = (unsigned*)(ws + alloc((size_t)TT * 4));
    int*      top2     = (int*)   (ws + alloc((size_t)TT * 2 * 4));
    int*      kept     = (int*)   (ws + alloc((size_t)EE * TT * 4));
    int*      slot_of  = (int*)   (ws + alloc((size_t)EE * TT * 4));
    int*      tok_of   = (int*)   (ws + alloc((size_t)EE * CAPP * 4));
    int*      nslots   = (int*)   (ws + alloc((size_t)EE * 4));
    __bf16*   xin      = (__bf16*)(ws + alloc((size_t)EE * CAPP * DD * 2));
    __bf16*   h1       = (__bf16*)(ws + alloc((size_t)EE * CAPP * DFFF * 2));
    float*    eout     = (float*) (ws + alloc((size_t)EE * CAPP * DD * 4));
    __bf16*   w1t      = (__bf16*)(ws + alloc((size_t)EE * DFFF * DD * 2));
    __bf16*   w2t      = (__bf16*)(ws + alloc((size_t)EE * DD * DFFF * 2));

    {
        long n = (long)VV * DD;
        convert_bf16_kernel<<<(unsigned)((n / 4 + 255) / 256), 256, 0, stream>>>(embed_W, embed_bf, n);
    }
    embed_gather_kernel<<<TT, 256, 0, stream>>>(ids, embed_W, h_a);

    float* hcur = h_a;
    float* hnext = h_b;
    for (int hop = 0; hop < 2; ++hop) {
        const float* Wr = routerW + (size_t)hop * DD * EE;
        const float* W1 = exW1 + (size_t)hop * EE * DD * DFFF;
        const float* W2 = exW2 + (size_t)hop * EE * DFFF * DD;

        router_kernel<<<TT / 256, 256, 0, stream>>>(hcur, Wr, logits, probs, maskbits, top2);
        cap_rank_kernel<<<dim3(TT / 256, EE), 256, 0, stream>>>(logits, maskbits, kept);
        cap_prefix_kernel<<<EE, 32, 0, stream>>>(kept, slot_of, tok_of, nslots);
        gather_kernel<<<EE * CAPP, 256, 0, stream>>>(hcur, tok_of, nslots, xin);

        transpose_convert_kernel<<<dim3(DD / 32, DFFF / 32, EE), dim3(32, 8), 0, stream>>>(W1, w1t, DD, DFFF);
        transpose_convert_kernel<<<dim3(DFFF / 32, DD / 32, EE), dim3(32, 8), 0, stream>>>(W2, w2t, DFFF, DD);

        gemm1_gelu_kernel<<<dim3(CAPP / 16, DFFF / 256, EE), 128, 0, stream>>>(xin, w1t, h1);
        gemm2_kernel<<<dim3(CAPP / 16, DD / 256, EE), 128, 0, stream>>>(h1, w2t, eout);

        combine_kernel<<<TT, 256, 0, stream>>>(hcur, probs, top2, kept, slot_of, eout, hnext);
        float* tmp = hcur; hcur = hnext; hnext = tmp;
    }

    rmsnorm_kernel<<<TT, 256, 0, stream>>>(hcur, ln_s, hn_bf);
    unembed_kernel<<<dim3(TT / 16, VV / 256), 128, 0, stream>>>(hn_bf, embed_bf, out);
}
